// encoderLayer_4904852652883
// MI455X (gfx1250) — matrix operand order repaired and checked by reading, not compiled, not run on hardware
//
#include <hip/hip_runtime.h>
#include <math.h>

// ---------------- types / WMMA helpers ----------------
typedef __bf16 v16bf __attribute__((ext_vector_type(16)));
typedef float  v8f   __attribute__((ext_vector_type(8)));
typedef unsigned int u32x4 __attribute__((ext_vector_type(4)));
typedef int i32x4 __attribute__((ext_vector_type(4)));
typedef int i32x8 __attribute__((ext_vector_type(8)));

union Frag {
    v16bf v;
    uint4 q[2];
};

__device__ __forceinline__ v8f wmma_bf16(v16bf a, v16bf b, v8f c) {
    // D = A(16x32 bf16) * B(32x16 bf16) + C(16x16 f32)
    return __builtin_amdgcn_wmma_f32_16x16x32_bf16(
        false, a, false, b, (short)0, c, false, false);
}

// DPP16 cross-lane move (VALU, no LDS traffic)
template <int CTRL>
__device__ __forceinline__ float dppf(float x) {
    return __builtin_bit_cast(float,
        __builtin_amdgcn_update_dpp(0, __builtin_bit_cast(int, x),
                                    CTRL, 0xF, 0xF, true));
}
// 16-lane butterfly reductions: quad_perm xor1, xor2, half_mirror, mirror
__device__ __forceinline__ float red16_max(float v) {
    v = fmaxf(v, dppf<0xB1>(v));
    v = fmaxf(v, dppf<0x4E>(v));
    v = fmaxf(v, dppf<0x141>(v));
    v = fmaxf(v, dppf<0x140>(v));
    return v;
}
__device__ __forceinline__ float red16_sum(float v) {
    v += dppf<0xB1>(v);
    v += dppf<0x4E>(v);
    v += dppf<0x141>(v);
    v += dppf<0x140>(v);
    return v;
}

// ---------------- problem constants ----------------
#define BB 4
#define SS 2048
#define EE 512
#define HH 8
#define DH 64
#define FFD 2048
#define MM (BB * SS)   // 8192 rows

// ---------------- weight convert + transpose: wt[n*K+k] = (bf16)w[k*N+n] ----
__global__ void convT(const float* __restrict__ w, __bf16* __restrict__ wt,
                      int K, int N) {
    int idx = blockIdx.x * 256 + threadIdx.x;
    if (idx >= K * N) return;
    int k = idx / N, n = idx % N;
    wt[(size_t)n * K + k] = (__bf16)w[idx];
}

// ---------------- pre-norm (mean, unbiased std, eps on std) -> bf16 --------
__global__ void norm_bf16(const float* __restrict__ x,
                          const float* __restrict__ alpha,
                          const float* __restrict__ beta,
                          __bf16* __restrict__ out) {
    __shared__ float s1[256], s2[256];
    int row = blockIdx.x, tid = threadIdx.x;
    const float* xr = x + (size_t)row * EE;
    float a = 0.f, b = 0.f;
    for (int i = tid; i < EE; i += 256) { float v = xr[i]; a += v; b += v * v; }
    s1[tid] = a; s2[tid] = b;
    __syncthreads();
    for (int st = 128; st > 0; st >>= 1) {
        if (tid < st) { s1[tid] += s1[tid + st]; s2[tid] += s2[tid + st]; }
        __syncthreads();
    }
    float mean = s1[0] / (float)EE;
    float var  = (s2[0] - (float)EE * mean * mean) / (float)(EE - 1);
    float sd   = sqrtf(fmaxf(var, 0.f));
    float sc   = alpha[0] / (sd + 1e-6f);
    float bt   = beta[0];
    __bf16* orow = out + (size_t)row * EE;
    for (int i = tid; i < EE; i += 256)
        orow[i] = (__bf16)((xr[i] - mean) * sc + bt);
}

// ---------------- TDM: stage 32 x K bf16 tile (global -> LDS) --------------
__device__ __forceinline__ void tdm_load_tile32xK(const __bf16* gsrc,
                                                  __bf16* lds, int K) {
    // LDS aperture: low 32 bits of generic pointer == LDS byte offset
    unsigned ldsoff = (unsigned)(uintptr_t)lds;
    unsigned long long ga = (unsigned long long)(uintptr_t)gsrc;
    unsigned d0 = (unsigned)(K / 2);        // row length in 4B units
    u32x4 g0;
    g0[0] = 1u;                             // count=1, user mode
    g0[1] = ldsoff;                         // lds_addr
    g0[2] = (unsigned)ga;                   // global_addr[31:0]
    g0[3] = (unsigned)((ga >> 32) & 0x1FFFFFFu) | (2u << 30); // addr[56:32]|type=2
    i32x8 g1;
    g1[0] = (int)(2u << 16);                // workgroup_mask=0, data_size=4B
    g1[1] = (int)((d0 & 0xFFFFu) << 16);    // tensor_dim0[15:0] @ bit48
    g1[2] = (int)((d0 >> 16) | (32u << 16)); // tensor_dim0[31:16] | tensor_dim1.lo=32
    g1[3] = (int)((d0 & 0xFFFFu) << 16);    // tensor_dim1.hi=0 | tile_dim0
    g1[4] = (int)32u;                       // tile_dim1=32, tile_dim2=0
    g1[5] = (int)d0;                        // tensor_dim0_stride lo
    g1[6] = 0;                              // stride hi / dim1_stride lo
    g1[7] = 0;
    i32x4 z4 = {0, 0, 0, 0};
#if defined(__clang_major__) && (__clang_major__ >= 23)
    i32x8 z8 = {0, 0, 0, 0, 0, 0, 0, 0};
    __builtin_amdgcn_tensor_load_to_lds(g0, g1, z4, z4, z8, 0);
#else
    __builtin_amdgcn_tensor_load_to_lds(g0, g1, z4, z4, 0);
#endif
    __builtin_amdgcn_s_wait_tensorcnt(0);
}

// ---------------- tiled WMMA GEMM ------------------------------------------
// C[M,N] = act((A[M,K]bf16 * Wt[N,K]^T + bias) * oscale); 4 waves/block;
// wave tile 32x64 (2 A-frags x 4 B-frags -> 8 WMMAs per 32-deep k-step);
// A block tile (32 x K) staged via Tensor Data Mover into dynamic LDS.
template <int RELU, int RESID, int BF16OUT, int TRANS>
__global__ void __launch_bounds__(128, 1)
gemm_ws(const __bf16* __restrict__ A,
        const __bf16* __restrict__ Wt,
        const float*  __restrict__ bias,
        const float*  __restrict__ resid,
        float*  __restrict__ outF,
        __bf16* __restrict__ outB,
        int N, int K, int Sdim, float oscale) {
    extern __shared__ __bf16 ldsA[];                 // 32 x K bf16
    const int wave = threadIdx.x >> 5, lane = threadIdx.x & 31;
    const int half = lane >> 4, l16 = lane & 15;
    const int m0 = blockIdx.x * 32;
    const int n0 = blockIdx.y * 256 + wave * 64;

    if (wave == 0) tdm_load_tile32xK(A + (size_t)m0 * K, ldsA, K);
    __syncthreads();

    const __bf16* a0 = ldsA + (size_t)l16 * K;
    const __bf16* a1 = ldsA + (size_t)(16 + l16) * K;
    const __bf16* wrow = Wt + (size_t)(n0 + l16) * K;

    v8f acc[2][4] = {};
    for (int kk = 0; kk < K; kk += 32) {
        Frag af0, af1;
        // A 16x32 layout: lane<16 K {kk..+7, kk+16..+23}; lane>=16 offset +8
        af0.q[0] = *(const uint4*)(a0 + kk + half * 8);
        af0.q[1] = *(const uint4*)(a0 + kk + 16 + half * 8);
        af1.q[0] = *(const uint4*)(a1 + kk + half * 8);
        af1.q[1] = *(const uint4*)(a1 + kk + 16 + half * 8);
        #pragma unroll
        for (int j = 0; j < 4; ++j) {
            const __bf16* wp = wrow + (size_t)j * 16 * K;
            Frag bf;
            // B 32x16 layout: col=l16, K {kk..+7, kk+16..+23}; lane>=16 offset +8
            bf.q[0] = *(const uint4*)(wp + kk + half * 8);
            bf.q[1] = *(const uint4*)(wp + kk + 16 + half * 8);
            acc[0][j] = wmma_bf16(af0.v, bf.v, acc[0][j]);
            acc[1][j] = wmma_bf16(af1.v, bf.v, acc[1][j]);
        }
    }

    #pragma unroll
    for (int j = 0; j < 4; ++j) {
        const int col = n0 + j * 16 + l16;
        const float bv = bias[col];
        #pragma unroll
        for (int i = 0; i < 2; ++i) {
            #pragma unroll
            for (int r = 0; r < 8; ++r) {
                int row = m0 + i * 16 + r + 8 * half;   // C layout rows r / r+8
                float val = (acc[i][j][r] + bv) * oscale;
                if (RELU)  val = fmaxf(val, 0.f);
                if (RESID) val += resid[(size_t)row * N + col];
                if (BF16OUT) {
                    if (TRANS) {  // store V as [B, E, S]
                        int bb = row / Sdim, s = row % Sdim;
                        outB[(size_t)bb * N * Sdim + (size_t)col * Sdim + s] =
                            (__bf16)val;
                    } else {
                        outB[(size_t)row * N + col] = (__bf16)val;
                    }
                } else {
                    outF[(size_t)row * N + col] = val;
                }
            }
        }
    }
}

// ---------------- flash attention: one wave = 16 query rows of one (b,h) ----
// Q is pre-scaled by 1/sqrt(DH) during its projection GEMM.
__global__ void __launch_bounds__(256, 1)
attn_kernel(const __bf16* __restrict__ qb,
            const __bf16* __restrict__ kb,
            const __bf16* __restrict__ vT,   // [B, E, S]
            const int*    __restrict__ mask, // [B, S]
            __bf16* __restrict__ ctx) {      // [B, S, E]
    const int wave = threadIdx.x >> 5, lane = threadIdx.x & 31;
    const int half = lane >> 4, l16 = lane & 15;
    const int gw = blockIdx.x * (blockDim.x >> 5) + wave;
    const int qt = gw & (SS / 16 - 1);
    const int h  = (gw >> 7) & (HH - 1);
    const int b  = gw >> 10;
    const int q0 = qt * 16;

    __shared__ __bf16 ldsP[8][16 * 32];
    __bf16* P = ldsP[wave];

    // Q fragments (A-layout), K dim = head dim (64) -> two 16x32 fragments
    const __bf16* qrow = qb + ((size_t)(b * SS + q0 + l16)) * EE + h * DH;
    Frag qa[2];
    #pragma unroll
    for (int s = 0; s < 2; ++s) {
        qa[s].q[0] = *(const uint4*)(qrow + s * 32 + half * 8);
        qa[s].q[1] = *(const uint4*)(qrow + s * 32 + 16 + half * 8);
    }

    float mrow[8], lrow[8];
    v8f o[4] = {};
    #pragma unroll
    for (int r = 0; r < 8; ++r) { mrow[r] = -3.0e38f; lrow[r] = 0.f; }

    for (int k0 = 0; k0 < SS; k0 += 32) {
        v8f sc[2];
        float madd[2];
        #pragma unroll
        for (int t = 0; t < 2; ++t) {
            int key = k0 + t * 16 + l16;                // this lane's column
            madd[t] = (mask[b * SS + key] == 0) ? -1e9f : 0.f;
            const __bf16* krow = kb + ((size_t)(b * SS + key)) * EE + h * DH;
            v8f a = {};
            #pragma unroll
            for (int s = 0; s < 2; ++s) {
                Frag kf;  // B-frag of K^T: col=key, K-dim = head-dim slice
                kf.q[0] = *(const uint4*)(krow + s * 32 + half * 8);
                kf.q[1] = *(const uint4*)(krow + s * 32 + 16 + half * 8);
                a = wmma_bf16(qa[s].v, kf.v, a);
            }
            sc[t] = a;
        }
        // online softmax over this 32-key tile (DPP reductions, no LDS)
        float al[8];
        #pragma unroll
        for (int r = 0; r < 8; ++r) {
            float v0 = sc[0][r] + madd[0];
            float v1 = sc[1][r] + madd[1];
            float rmax = red16_max(fmaxf(v0, v1));
            float mnew = fmaxf(mrow[r], rmax);
            al[r] = __expf(mrow[r] - mnew);
            mrow[r] = mnew;
            float p0 = __expf(v0 - mnew);
            float p1 = __expf(v1 - mnew);
            // C-layout row of this lane: r + 8*half ; columns l16 / 16+l16
            P[(r + 8 * half) * 32 + l16]      = (__bf16)p0;
            P[(r + 8 * half) * 32 + 16 + l16] = (__bf16)p1;
            lrow[r] = lrow[r] * al[r] + red16_sum(p0 + p1);
        }
        #pragma unroll
        for (int g = 0; g < 4; ++g)
            #pragma unroll
            for (int r = 0; r < 8; ++r) o[g][r] *= al[r];

        asm volatile("s_wait_dscnt 0x0" ::: "memory");  // wave-local LDS RAW

        // re-read P as A-fragment (16 q rows x 32 keys)
        Frag pf;
        pf.q[0] = *(const uint4*)(P + l16 * 32 + half * 8);
        pf.q[1] = *(const uint4*)(P + l16 * 32 + 16 + half * 8);

        // ctx += P * V ; V B-frags contiguous thanks to [B,E,S] layout
        #pragma unroll
        for (int g = 0; g < 4; ++g) {
            const __bf16* vrow = vT + (size_t)b * EE * SS
                               + (size_t)(h * DH + g * 16 + l16) * SS
                               + k0 + half * 8;
            Frag vf;
            vf.q[0] = *(const uint4*)(vrow);
            vf.q[1] = *(const uint4*)(vrow + 16);
            o[g] = wmma_bf16(pf.v, vf.v, o[g]);
        }
    }

    // normalize and store ctx (bf16) back in [B,S,E]
    #pragma unroll
    for (int r = 0; r < 8; ++r) {
        float inv = 1.f / lrow[r];
        int row = q0 + r + 8 * half;
        #pragma unroll
        for (int g = 0; g < 4; ++g)
            ctx[((size_t)(b * SS + row)) * EE + h * DH + g * 16 + l16] =
                (__bf16)(o[g][r] * inv);
    }
}

// ---------------- host orchestration ----------------
extern "C" void kernel_launch(void* const* d_in, const int* in_sizes, int n_in,
                              void* d_out, int out_size, void* d_ws, size_t ws_size,
                              hipStream_t stream) {
    (void)in_sizes; (void)n_in; (void)out_size; (void)ws_size;
    const float* x     = (const float*)d_in[0];
    const int*   smask = (const int*)  d_in[1];
    const float* wq = (const float*)d_in[2];  const float* bq = (const float*)d_in[3];
    const float* wk = (const float*)d_in[4];  const float* bk = (const float*)d_in[5];
    const float* wv = (const float*)d_in[6];  const float* bv = (const float*)d_in[7];
    const float* wo = (const float*)d_in[8];  const float* bo = (const float*)d_in[9];
    const float* w1 = (const float*)d_in[10]; const float* b1 = (const float*)d_in[11];
    const float* w2 = (const float*)d_in[12]; const float* b2 = (const float*)d_in[13];
    const float* alpha1 = (const float*)d_in[14]; const float* beta1 = (const float*)d_in[15];
    const float* alpha2 = (const float*)d_in[16]; const float* beta2 = (const float*)d_in[17];
    float* out = (float*)d_out;

    // workspace carve-up
    char* ws = (char*)d_ws;
    size_t off = 0;
    auto carve = [&](size_t bytes) { char* p = ws + off; off = (off + bytes + 255) & ~(size_t)255; return p; };
    __bf16* nbf  = (__bf16*)carve((size_t)MM * EE * 2);
    __bf16* qbf  = (__bf16*)carve((size_t)MM * EE * 2);
    __bf16* kbf  = (__bf16*)carve((size_t)MM * EE * 2);
    __bf16* vTb  = (__bf16*)carve((size_t)MM * EE * 2);   // [B,E,S]
    __bf16* ctx  = (__bf16*)carve((size_t)MM * EE * 2);
    float*  x2   = (float*) carve((size_t)MM * EE * 4);
    __bf16* n2   = (__bf16*)carve((size_t)MM * EE * 2);
    __bf16* ffb  = (__bf16*)carve((size_t)MM * FFD * 2);
    __bf16* wqT  = (__bf16*)carve((size_t)EE * EE * 2);
    __bf16* wkT  = (__bf16*)carve((size_t)EE * EE * 2);
    __bf16* wvT  = (__bf16*)carve((size_t)EE * EE * 2);
    __bf16* woT  = (__bf16*)carve((size_t)EE * EE * 2);
    __bf16* w1T  = (__bf16*)carve((size_t)EE * FFD * 2);
    __bf16* w2T  = (__bf16*)carve((size_t)FFD * EE * 2);

    // 1) weight convert+transpose (fp32 -> bf16, [K,N] -> [N,K])
    int eN = EE * EE, fN = EE * FFD;
    convT<<<(eN + 255) / 256, 256, 0, stream>>>(wq, wqT, EE, EE);
    convT<<<(eN + 255) / 256, 256, 0, stream>>>(wk, wkT, EE, EE);
    convT<<<(eN + 255) / 256, 256, 0, stream>>>(wv, wvT, EE, EE);
    convT<<<(eN + 255) / 256, 256, 0, stream>>>(wo, woT, EE, EE);
    convT<<<(fN + 255) / 256, 256, 0, stream>>>(w1, w1T, EE, FFD);
    convT<<<(fN + 255) / 256, 256, 0, stream>>>(w2, w2T, FFD, EE);

    // 2) pre-norm 1
    norm_bf16<<<MM, 256, 0, stream>>>(x, alpha1, beta1, nbf);

    // 3) QKV projections (Q pre-scaled by 1/sqrt(DH); V stored transposed)
    dim3 gE(MM / 32, EE / 256), blk(128);
    size_t shE = (size_t)32 * EE * 2;     // 32 KB
    size_t shF = (size_t)32 * FFD * 2;    // 128 KB
    gemm_ws<0,0,1,0><<<gE, blk, shE, stream>>>(nbf, wqT, bq, nullptr, nullptr, qbf, EE, EE, SS, 0.125f);
    gemm_ws<0,0,1,0><<<gE, blk, shE, stream>>>(nbf, wkT, bk, nullptr, nullptr, kbf, EE, EE, SS, 1.0f);
    gemm_ws<0,0,1,1><<<gE, blk, shE, stream>>>(nbf, wvT, bv, nullptr, nullptr, vTb, EE, EE, SS, 1.0f);

    // 4) flash attention (one wave per 16 query rows per head)
    int nWaves = BB * HH * (SS / 16);          // 4096 waves
    attn_kernel<<<nWaves / 8, 256, 0, stream>>>(qbf, kbf, vTb, smask, ctx);

    // 5) output projection + residual -> x2 (fp32)
    gemm_ws<0,1,0,0><<<gE, blk, shE, stream>>>(ctx, woT, bo, x, x2, nullptr, EE, EE, SS, 1.0f);

    // 6) pre-norm 2
    norm_bf16<<<MM, 256, 0, stream>>>(x2, alpha2, beta2, n2);

    // 7) FFN
    dim3 gF1(MM / 32, FFD / 256);
    gemm_ws<1,0,1,0><<<gF1, blk, shE, stream>>>(n2, w1T, b1, nullptr, nullptr, ffb, FFD, EE, SS, 1.0f);
    gemm_ws<0,1,0,0><<<gE,  blk, shF, stream>>>(ffb, w2T, b2, x2, out, nullptr, EE, FFD, SS, 1.0f);
}
